// RGCNAtt2Layer_33526514713112
// MI455X (gfx1250) — matrix operand order repaired and checked by reading, not compiled, not run on hardware
//
#include <hip/hip_runtime.h>

// ---------------- problem constants ----------------
#define N_NODES   50000
#define N_EDGES   800000
#define IN_FEAT   100
#define OUT_FEAT  100
#define NUM_RELS  200
#define NH_OUT    300     // NUM_HEADS * OUT_FEAT
#define KPAD      128     // K padded 100 -> 128 (4 chunks of 32)
#define KSTRIDE   136     // bf16 row stride (=272B) -> 4-bank rotation, no LDS conflicts
#define NPAD_FC   304     // 300 -> 19 tiles of 16
#define NPAD_LOOP 112     // 100 -> 7 tiles of 16
#define M_TILE    128
#define BLOCK     256     // 8 waves (wave32)
#define LDS_B_OFF 34816u  // byte offset of lB in dynamic LDS (= M_TILE*KSTRIDE*2), both kernels

// ---------------- WMMA types (gfx1250) ----------------
typedef __bf16 v16bf __attribute__((ext_vector_type(16)));
typedef float  v8f   __attribute__((ext_vector_type(8)));
union FragBF { v16bf v; uint4 q[2]; };

__device__ __forceinline__ unsigned short f2bf(float f) {
  union { float f; unsigned u; } c; c.f = f;
  unsigned u = c.u;
  return (unsigned short)((u + 0x7FFFu + ((u >> 16) & 1u)) >> 16); // RNE
}

// gfx1250 async global->LDS DMA (ASYNCcnt-tracked), per CDNA5 ISA.
// ldsoff = byte offset within this workgroup's LDS allocation (dynamic segment starts at 0).
__device__ __forceinline__ void async_g2l_b128(unsigned ldsoff, const void* gaddr) {
  unsigned long long ga = (unsigned long long)(uintptr_t)gaddr;
  asm volatile("global_load_async_to_lds_b128 %0, %1, off" :: "v"(ldsoff), "v"(ga) : "memory");
}
__device__ __forceinline__ void wait_asynccnt0() {
  asm volatile("s_wait_asynccnt 0x0" ::: "memory");
}

// ---------------- tiny prep kernels ----------------
__global__ void k_zero_f32(float* __restrict__ p, int n) {
  int i = blockIdx.x * blockDim.x + threadIdx.x;
  if (i < n) p[i] = 0.0f;
}

// fc_w (100 x 300, row-major [K][N]) -> bf16 [NPAD_FC][KSTRIDE] (transposed, zero-padded)
__global__ void k_conv_fc(const float* __restrict__ fc_w, unsigned short* __restrict__ B) {
  int i = blockIdx.x * blockDim.x + threadIdx.x;
  if (i >= NPAD_FC * KSTRIDE) return;
  int n = i / KSTRIDE, k = i - n * KSTRIDE;
  float v = (n < NH_OUT && k < IN_FEAT) ? fc_w[(size_t)k * NH_OUT + n] : 0.0f;
  B[i] = f2bf(v);
}

// loop_w (100 x 100) -> bf16 [NPAD_LOOP][KSTRIDE]
__global__ void k_conv_loop(const float* __restrict__ loop_w, unsigned short* __restrict__ B) {
  int i = blockIdx.x * blockDim.x + threadIdx.x;
  if (i >= NPAD_LOOP * KSTRIDE) return;
  int n = i / KSTRIDE, k = i - n * KSTRIDE;
  float v = (n < OUT_FEAT && k < IN_FEAT) ? loop_w[(size_t)k * OUT_FEAT + n] : 0.0f;
  B[i] = f2bf(v);
}

// ---------------- main edge kernel: fc GEMM (WMMA bf16) + fused attention + scatter ----------------
__global__ void __launch_bounds__(BLOCK)
k_edge(const float* __restrict__ e,
       const float* __restrict__ h,
       const float* __restrict__ weight,
       const float* __restrict__ attn,
       const int*   __restrict__ rel,
       const int*   __restrict__ src,
       const int*   __restrict__ dst,
       const unsigned short* __restrict__ Bfc,
       float* __restrict__ agg)
{
  extern __shared__ unsigned char smem[];
  unsigned short* lA  = (unsigned short*)smem;                 // [128][136] bf16 of e-tile
  unsigned short* lB  = lA + M_TILE * KSTRIDE;                 // [304][136] bf16 of fc_w^T
  int* lrel = (int*)(lB + NPAD_FC * KSTRIDE);
  int* lsrc = lrel + M_TILE;
  int* ldst = lsrc + M_TILE;

  const int tid = threadIdx.x;
  const int eb  = blockIdx.x * M_TILE;

  // kick off async DMA of the whole bf16 fc_w (82.7KB, L2-resident) straight into LDS
  for (int i = tid; i < (NPAD_FC * KSTRIDE * 2) / 16; i += BLOCK)
    async_g2l_b128(LDS_B_OFF + (unsigned)i * 16u, (const char*)Bfc + (size_t)i * 16u);

  // stage e-tile: flat-contiguous float4 stream (rows = 25 float4, never straddles a row)
  const float4* eq = (const float4*)(e + (size_t)eb * IN_FEAT);
  for (int i = tid; i < (M_TILE * IN_FEAT) / 4; i += BLOCK) {
    float4 v = eq[i];
    int f = i * 4;
    int r = f / IN_FEAT, c = f - r * IN_FEAT;
    unsigned p0 = (unsigned)f2bf(v.x) | ((unsigned)f2bf(v.y) << 16);
    unsigned p1 = (unsigned)f2bf(v.z) | ((unsigned)f2bf(v.w) << 16);
    *(uint2*)&lA[r * KSTRIDE + c] = make_uint2(p0, p1);
  }
  for (int i = tid; i < M_TILE * (KSTRIDE - IN_FEAT); i += BLOCK) {
    int r = i / (KSTRIDE - IN_FEAT), c = IN_FEAT + (i - r * (KSTRIDE - IN_FEAT));
    lA[r * KSTRIDE + c] = 0;
  }
  if (tid < M_TILE) {
    lrel[tid] = rel[eb + tid];
    lsrc[tid] = src[eb + tid];
    ldst[tid] = dst[eb + tid];
  }
  wait_asynccnt0();
  __syncthreads();

  // block-diagonal message (SUB_IN=SUB_OUT=1 => elementwise): agg[dst] += h[src]*weight[rel]
  for (int i = tid; i < M_TILE * OUT_FEAT; i += BLOCK) {
    int el = i / OUT_FEAT, o = i - el * OUT_FEAT;
    float v = h[(size_t)lsrc[el] * IN_FEAT + o] * weight[(size_t)lrel[el] * OUT_FEAT + o];
    unsafeAtomicAdd(&agg[(size_t)ldst[el] * OUT_FEAT + o], v);
  }

  // per-wave WMMA: one 16-edge M-subtile, sweep 19 N-tiles x 4 K-chunks
  const int wv = tid >> 5, ln = tid & 31;
  const int kh = ln >> 4, lno = ln & 15;

  FragBF a[4];                                  // A fragments: contiguous K order, k = 16*kh + s
  const int arow = (wv * 16 + lno) * KSTRIDE;
  #pragma unroll
  for (int kc = 0; kc < 4; ++kc) {
    const int kb = kc * 32 + kh * 16;           // lane-half holds 16 consecutive K (32B)
    a[kc].q[0] = *(const uint4*)&lA[arow + kb];
    a[kc].q[1] = *(const uint4*)&lA[arow + kb + 8];
  }

  for (int nt = 0; nt < NPAD_FC / 16; ++nt) {
    const int brow = (nt * 16 + lno) * KSTRIDE;
    FragBF b[4];                                // hoist all B frags: loads drain under WMMA chain
    #pragma unroll
    for (int kc = 0; kc < 4; ++kc) {
      const int kb = kc * 32 + kh * 16;         // lane-half holds 16 consecutive K (32B)
      b[kc].q[0] = *(const uint4*)&lB[brow + kb];
      b[kc].q[1] = *(const uint4*)&lB[brow + kb + 8];
    }
    v8f acc = {};
    #pragma unroll
    for (int kc = 0; kc < 4; ++kc)
      acc = __builtin_amdgcn_wmma_f32_16x16x32_bf16(false, a[kc].v, false, b[kc].v,
                                                    (short)0, acc, false, false);
    // epilogue: C layout -> lane = N, vgpr r -> M = 8*kh + r
    const int ncol = nt * 16 + lno;             // head*100 + out  (attn row-major matches)
    if (ncol < NH_OUT) {
      int o = ncol; if (o >= 200) o -= 200; else if (o >= 100) o -= 100;
      #pragma unroll
      for (int r = 0; r < 8; ++r) {
        const int el = wv * 16 + kh * 8 + r;
        float t = attn[(size_t)lrel[el] * NH_OUT + ncol] * acc[r];
        t = (t > 0.0f) ? t : 0.2f * t;          // leaky_relu(a*fc, 0.2)
        unsafeAtomicAdd(&agg[(size_t)ldst[el] * OUT_FEAT + o], t);
      }
    }
  }
}

// ---------------- final: out = agg*norm + h @ loop_w (WMMA) ----------------
__global__ void __launch_bounds__(BLOCK)
k_final(const float* __restrict__ h,
        const float* __restrict__ norm,
        const float* __restrict__ agg,
        const unsigned short* __restrict__ Bloop,
        float* __restrict__ out)
{
  extern __shared__ unsigned char smem[];
  unsigned short* lA = (unsigned short*)smem;          // [128][136] bf16 of h-tile
  unsigned short* lB = lA + M_TILE * KSTRIDE;          // [112][136] bf16 of loop_w^T

  const int tid = threadIdx.x;
  const int nb  = blockIdx.x * M_TILE;

  for (int i = tid; i < (NPAD_LOOP * KSTRIDE * 2) / 16; i += BLOCK)
    async_g2l_b128(LDS_B_OFF + (unsigned)i * 16u, (const char*)Bloop + (size_t)i * 16u);

  for (int i = tid; i < (M_TILE * IN_FEAT) / 4; i += BLOCK) {
    int f = i * 4;
    int r = f / IN_FEAT, c = f - r * IN_FEAT;
    int node = nb + r;
    float4 v = (node < N_NODES) ? *(const float4*)&h[(size_t)node * IN_FEAT + c]
                                : make_float4(0.f, 0.f, 0.f, 0.f);
    unsigned p0 = (unsigned)f2bf(v.x) | ((unsigned)f2bf(v.y) << 16);
    unsigned p1 = (unsigned)f2bf(v.z) | ((unsigned)f2bf(v.w) << 16);
    *(uint2*)&lA[r * KSTRIDE + c] = make_uint2(p0, p1);
  }
  for (int i = tid; i < M_TILE * (KSTRIDE - IN_FEAT); i += BLOCK) {
    int r = i / (KSTRIDE - IN_FEAT), c = IN_FEAT + (i - r * (KSTRIDE - IN_FEAT));
    lA[r * KSTRIDE + c] = 0;
  }
  wait_asynccnt0();
  __syncthreads();

  const int wv = tid >> 5, ln = tid & 31;
  const int kh = ln >> 4, lno = ln & 15;

  FragBF a[4];
  const int arow = (wv * 16 + lno) * KSTRIDE;
  #pragma unroll
  for (int kc = 0; kc < 4; ++kc) {
    const int kb = kc * 32 + kh * 16;
    a[kc].q[0] = *(const uint4*)&lA[arow + kb];
    a[kc].q[1] = *(const uint4*)&lA[arow + kb + 8];
  }

  for (int nt = 0; nt < NPAD_LOOP / 16; ++nt) {
    const int brow = (nt * 16 + lno) * KSTRIDE;
    FragBF b[4];
    #pragma unroll
    for (int kc = 0; kc < 4; ++kc) {
      const int kb = kc * 32 + kh * 16;
      b[kc].q[0] = *(const uint4*)&lB[brow + kb];
      b[kc].q[1] = *(const uint4*)&lB[brow + kb + 8];
    }
    v8f acc = {};
    #pragma unroll
    for (int kc = 0; kc < 4; ++kc)
      acc = __builtin_amdgcn_wmma_f32_16x16x32_bf16(false, a[kc].v, false, b[kc].v,
                                                    (short)0, acc, false, false);
    const int ncol = nt * 16 + lno;
    if (ncol < OUT_FEAT) {
      #pragma unroll
      for (int r = 0; r < 8; ++r) {
        const int node = nb + wv * 16 + kh * 8 + r;
        if (node < N_NODES) {
          const size_t idx = (size_t)node * OUT_FEAT + ncol;
          out[idx] = acc[r] + agg[idx] * norm[node];
        }
      }
    }
  }
}

// ---------------- launcher ----------------
extern "C" void kernel_launch(void* const* d_in, const int* in_sizes, int n_in,
                              void* d_out, int out_size, void* d_ws, size_t ws_size,
                              hipStream_t stream) {
  (void)in_sizes; (void)n_in; (void)out_size;
  const float* h      = (const float*)d_in[0];
  const float* e      = (const float*)d_in[1];
  const float* norm   = (const float*)d_in[2];
  const float* weight = (const float*)d_in[3];
  const float* attn   = (const float*)d_in[4];
  const float* fc_w   = (const float*)d_in[5];
  const float* loop_w = (const float*)d_in[6];
  const int*   rel    = (const int*)d_in[7];
  const int*   src    = (const int*)d_in[8];
  const int*   dst    = (const int*)d_in[9];
  float* out = (float*)d_out;

  // workspace layout (all offsets 16B aligned)
  const size_t AGG_BYTES   = (size_t)N_NODES * OUT_FEAT * 4;            // 20,000,000
  const size_t BFC_BYTES   = (size_t)NPAD_FC * KSTRIDE * 2;             // 82,688
  const size_t BLOOP_BYTES = (size_t)NPAD_LOOP * KSTRIDE * 2;           // 30,464
  if (ws_size < AGG_BYTES + BFC_BYTES + BLOOP_BYTES) return;
  char* ws = (char*)d_ws;
  float*          agg   = (float*)ws;
  unsigned short* Bfc   = (unsigned short*)(ws + AGG_BYTES);
  unsigned short* Bloop = (unsigned short*)(ws + AGG_BYTES + BFC_BYTES);

  const int aggN = N_NODES * OUT_FEAT;
  k_zero_f32 <<<(aggN + 255) / 256, 256, 0, stream>>>(agg, aggN);
  k_conv_fc  <<<(NPAD_FC * KSTRIDE + 255) / 256, 256, 0, stream>>>(fc_w, Bfc);
  k_conv_loop<<<(NPAD_LOOP * KSTRIDE + 255) / 256, 256, 0, stream>>>(loop_w, Bloop);

  const size_t edge_smem  = (size_t)M_TILE * KSTRIDE * 2 + (size_t)NPAD_FC * KSTRIDE * 2
                          + 3 * M_TILE * sizeof(int);                   // 119,040 B
  const size_t final_smem = (size_t)M_TILE * KSTRIDE * 2 + (size_t)NPAD_LOOP * KSTRIDE * 2; // 65,280 B

  k_edge <<<N_EDGES / M_TILE, BLOCK, edge_smem, stream>>>(e, h, weight, attn, rel, src, dst, Bfc, agg);
  k_final<<<(N_NODES + M_TILE - 1) / M_TILE, BLOCK, final_smem, stream>>>(h, norm, agg, Bloop, out);
}
